// MoE_29334626632568
// MI455X (gfx1250) — matrix operand order repaired and checked by reading, not compiled, not run on hardware
//
#include <hip/hip_runtime.h>
#include <hip/hip_bf16.h>
#include <math.h>

// ---------------- problem constants ----------------
#define T_TOK 8192      // B*S tokens
#define DIM   1024      // hidden
#define NEXP  16        // routed experts
#define TOPK  2
#define IDIM  1024      // routed expert inter dim
#define SHI   2048      // shared expert inter dim
#define ROWS  32        // token tile rows per workgroup
#define LSTR  (DIM + 8) // LDS row stride (bf16 elems)

typedef __bf16 bf16;
typedef bf16  v16bf __attribute__((ext_vector_type(16)));
typedef bf16  v8bf  __attribute__((ext_vector_type(8)));
typedef float v8f   __attribute__((ext_vector_type(8)));

// f32 -> bf16, round-to-nearest-even
static __device__ __forceinline__ bf16 f2bf(float f) {
  unsigned int u = __builtin_bit_cast(unsigned int, f);
  u += 0x7FFFu + ((u >> 16) & 1u);
  unsigned short h = (unsigned short)(u >> 16);
  return __builtin_bit_cast(bf16, h);
}

static __device__ __forceinline__ v16bf cat8(v8bf lo, v8bf hi) {
  return __builtin_shufflevector(lo, hi, 0,1,2,3,4,5,6,7,8,9,10,11,12,13,14,15);
}

// gfx1250 async global->LDS copy (16B per lane), tracked by ASYNCcnt.
// lds_byte_off is the byte offset within the workgroup's LDS allocation
// (kernels below use only dynamic LDS, which starts at offset 0).
static __device__ __forceinline__ void async_b128_to_lds(unsigned lds_byte_off,
                                                         const void* gsrc) {
  asm volatile("global_load_async_to_lds_b128 %0, %1, off"
               :: "v"(lds_byte_off), "v"(gsrc)
               : "memory");
}
static __device__ __forceinline__ void wait_async_all() {
  asm volatile("s_wait_asynccnt 0x0" ::: "memory");
}

// A fragment (16x32 bf16, M x K) from LDS row-major tile.
// lanes 0-15 row m=lane hold k={0..7,16..23}; lanes 16-31 row m=lane-16 hold k={8..15,24..31}.
static __device__ __forceinline__ v16bf load_a_frag(const bf16* tile, int stride,
                                                    int rowbase, int k0, int lane) {
  const int m    = rowbase + (lane & 15);
  const int koff = (lane >> 4) << 3;             // 0 or 8
  const bf16* p  = tile + m * stride + k0 + koff;
  v8bf lo = *(const v8bf*)p;                     // k0+koff .. +7
  v8bf hi = *(const v8bf*)(p + 16);              // k0+16+koff .. +7
  return cat8(lo, hi);
}

// B fragment from pre-swizzled bf16 weights: fragment (ntile, kc) is 1KB contiguous,
// lane L owns 16 consecutive bf16 at +L*16 (exact WMMA B register image).
static __device__ __forceinline__ v16bf load_b_tiled(const bf16* Wt, int kchunks,
                                                     int ntile, int kc, int lane) {
  const bf16* p = Wt + ((((size_t)ntile * kchunks + kc) << 9) | ((size_t)lane << 4));
  v8bf lo = *(const v8bf*)p;
  v8bf hi = *(const v8bf*)(p + 8);
  return cat8(lo, hi);
}

#define WMMA_BF16(a, b, c) \
  __builtin_amdgcn_wmma_f32_16x16x32_bf16(false, (a), false, (b), (short)0, (c), false, false)

// ---------------- kernel 0: zero per-expert state ----------------
__global__ void moe_init_kernel(int* __restrict__ cnt, float* __restrict__ psum) {
  int i = threadIdx.x;
  if (i < NEXP) { cnt[i] = 0; psum[i] = 0.f; }
}

// ---------------- weight converter: f32 row-major [O,K] -> bf16 fragment-tiled ----------------
__global__ void __launch_bounds__(256)
convert_w_kernel(const float* __restrict__ src, bf16* __restrict__ dst,
                 int K, int ksh, long total8) {
  long idx = (long)blockIdx.x * 256 + threadIdx.x;
  if (idx >= total8) return;
  long f   = idx >> 6;                       // fragment (512 elems = 64 8-groups)
  int  g   = (int)(idx & 63);
  int  lane = g >> 1, h = g & 1;
  long nt  = f >> ksh;
  int  kc  = (int)(f & ((1 << ksh) - 1));
  int  row = (int)(nt * 16 + (lane & 15));
  int  col = kc * 32 + ((lane >> 4) << 3) + h * 16;
  const float* p = src + (size_t)row * K + col;
  float4 a = *(const float4*)p;
  float4 b = *(const float4*)(p + 4);
  v8bf v;
  v[0]=f2bf(a.x); v[1]=f2bf(a.y); v[2]=f2bf(a.z); v[3]=f2bf(a.w);
  v[4]=f2bf(b.x); v[5]=f2bf(b.y); v[6]=f2bf(b.z); v[7]=f2bf(b.w);
  *(v8bf*)(dst + ((size_t)idx << 3)) = v;
}

// ---------------- x converter: f32 -> bf16 row-major ----------------
__global__ void __launch_bounds__(256)
convert_x_kernel(const float* __restrict__ src, bf16* __restrict__ dst) {
  size_t idx = ((size_t)blockIdx.x * 256 + threadIdx.x);   // per 8 elems
  const float* p = src + (idx << 3);
  float4 a = *(const float4*)p;
  float4 b = *(const float4*)(p + 4);
  v8bf v;
  v[0]=f2bf(a.x); v[1]=f2bf(a.y); v[2]=f2bf(a.z); v[3]=f2bf(a.w);
  v[4]=f2bf(b.x); v[5]=f2bf(b.y); v[6]=f2bf(b.z); v[7]=f2bf(b.w);
  *(v8bf*)(dst + (idx << 3)) = v;
}

// ---------------- gate: sigmoid scores, top-2, expert token lists ----------------
__global__ void __launch_bounds__(256)
moe_gate_kernel(const float* __restrict__ x, const float* __restrict__ gw,
                int* __restrict__ cnt, float* __restrict__ psum,
                int* __restrict__ list, float* __restrict__ wlist) {
  const int t = blockIdx.x * 256 + threadIdx.x;
  if (t >= T_TOK) return;
  float acc[NEXP];
#pragma unroll
  for (int e = 0; e < NEXP; ++e) acc[e] = 0.f;
  const float* xr = x + (size_t)t * DIM;
  for (int d = 0; d < DIM; d += 4) {
    float4 xv = *(const float4*)(xr + d);
#pragma unroll
    for (int e = 0; e < NEXP; ++e) {
      const float* g = gw + (size_t)e * DIM + d;   // uniform -> scalar loads
      acc[e] += xv.x * g[0] + xv.y * g[1] + xv.z * g[2] + xv.w * g[3];
    }
  }
  float sc[NEXP];
#pragma unroll
  for (int e = 0; e < NEXP; ++e) sc[e] = 1.f / (1.f + __expf(-acc[e]));
  int e0 = 0; float v0 = sc[0];
#pragma unroll
  for (int e = 1; e < NEXP; ++e) if (sc[e] > v0) { v0 = sc[e]; e0 = e; }
  int e1 = (e0 == 0) ? 1 : 0; float v1 = sc[e1];
#pragma unroll
  for (int e = 0; e < NEXP; ++e) if (e != e0 && sc[e] > v1) { v1 = sc[e]; e1 = e; }
  const float inv = 1.f / (v0 + v1);
  const float w0 = v0 * inv, w1 = v1 * inv;

  int p0 = atomicAdd(cnt + e0, 1);
  list[(size_t)e0 * T_TOK + p0]  = t;
  wlist[(size_t)e0 * T_TOK + p0] = w0;
  atomicAdd(psum + e0, w0);
  int p1 = atomicAdd(cnt + e1, 1);
  list[(size_t)e1 * T_TOK + p1]  = t;
  wlist[(size_t)e1 * T_TOK + p1] = w1;
  atomicAdd(psum + e1, w1);
}

// ---------------- aux loss ----------------
__global__ void moe_aux_kernel(const int* __restrict__ cnt,
                               const float* __restrict__ psum,
                               float* __restrict__ outL) {
  if (threadIdx.x == 0) {
    float L = 0.f;
    for (int e = 0; e < NEXP; ++e) {
      float f = (float)NEXP * (float)cnt[e] / (float)(TOPK * T_TOK);
      float P = psum[e] / (float)T_TOK;
      L += f * P;
    }
    *outL = L;
  }
}

// ---------------- shared expert: dense SwiGLU with bias, writes z ----------------
__global__ void __launch_bounds__(256)
moe_shared_kernel(const bf16* __restrict__ xb,
                  const bf16* __restrict__ sw1b, const float* __restrict__ sb1,
                  const bf16* __restrict__ sw2b, const float* __restrict__ sb2,
                  const bf16* __restrict__ sw3b, const float* __restrict__ sb3,
                  float* __restrict__ out) {
  const int tilebase = blockIdx.x * ROWS;
  extern __shared__ char smem[];
  bf16* xs = (bf16*)smem;                                   // ROWS x LSTR @ LDS offset 0
  bf16* hs = (bf16*)(smem + (size_t)ROWS * LSTR * sizeof(bf16));

  const int tid = threadIdx.x, lane = tid & 31, wid = tid >> 5;
  const int rb = wid & 1, cg = wid >> 1;                    // row-block 0/1, col-group 0..3
  const int rowbase = rb * 16;

  // stage 0: async-copy bf16 x tile -> LDS (global_load_async_to_lds_b128)
  for (int idx = tid; idx < ROWS * (DIM / 8); idx += 256) {
    int m  = idx >> 7;                                      // DIM/8 = 128 per row
    int d8 = (idx & 127) << 3;
    async_b128_to_lds((unsigned)(m * LSTR + d8) * 2u,
                      xb + (size_t)(tilebase + m) * DIM + d8);
  }
  wait_async_all();
  __syncthreads();

  const v8f vz = {0.f,0.f,0.f,0.f,0.f,0.f,0.f,0.f};
  v8f acc[16];
#pragma unroll
  for (int nt = 0; nt < 16; ++nt) acc[nt] = vz;

  for (int half = 0; half < 2; ++half) {
    // ---- stage 1: h_half = silu(x sw1^T + b1) * (x sw3^T + b3) -> LDS ----
#pragma unroll 1
    for (int nt = 0; nt < 16; ++nt) {
      const int ncl_loc = cg * 256 + nt * 16;               // 0..1023 within half
      const int ntile_g = half * 64 + cg * 16 + nt;         // global inter ntile (SHI/16 = 128)
      v8f c1 = vz, c3 = vz;
#pragma unroll 2
      for (int kc = 0; kc < DIM / 32; ++kc) {
        v16bf a  = load_a_frag(xs, LSTR, rowbase, kc << 5, lane);
        v16bf b1 = load_b_tiled(sw1b, DIM / 32, ntile_g, kc, lane);
        v16bf b3 = load_b_tiled(sw3b, DIM / 32, ntile_g, kc, lane);
        c1 = WMMA_BF16(a, b1, c1);
        c3 = WMMA_BF16(a, b3, c3);
      }
      const int mbase = rowbase + ((lane >> 4) << 3);
      const int nloc  = lane & 15;
      const float b1v = sb1[ntile_g * 16 + nloc];
      const float b3v = sb3[ntile_g * 16 + nloc];
#pragma unroll
      for (int r = 0; r < 8; ++r) {
        float s = c1[r] + b1v;
        float h = (s / (1.f + __expf(-s))) * (c3[r] + b3v);
        hs[(mbase + r) * LSTR + ncl_loc + nloc] = f2bf(h);
      }
    }
    __syncthreads();

    // ---- stage 2: acc += h_half @ sw2^T (K range of this half) ----
#pragma unroll
    for (int nt = 0; nt < 16; ++nt) {
      const int ntile = cg * 16 + nt;                       // output ntile in D
      v8f c = acc[nt];
#pragma unroll 2
      for (int kc = 0; kc < 32; ++kc) {
        v16bf a = load_a_frag(hs, LSTR, rowbase, kc << 5, lane);
        v16bf b = load_b_tiled(sw2b, SHI / 32, ntile, half * 32 + kc, lane);
        c = WMMA_BF16(a, b, c);
      }
      acc[nt] = c;
    }
    __syncthreads();                                        // before hs overwritten
  }

  // epilogue: z = acc + b2, plain store (routed kernel atomic-adds on top)
  const int mbase = rowbase + ((lane >> 4) << 3);
#pragma unroll
  for (int nt = 0; nt < 16; ++nt) {
    const int n = cg * 256 + nt * 16 + (lane & 15);
    const float b2 = sb2[n];
#pragma unroll
    for (int r = 0; r < 8; ++r)
      out[(size_t)(tilebase + mbase + r) * DIM + n] = acc[nt][r] + b2;
  }
}

// ---------------- routed experts: gathered SwiGLU, weighted atomic combine ----------------
__global__ void __launch_bounds__(256)
moe_routed_kernel(const bf16* __restrict__ xb,
                  const bf16* __restrict__ w1b, const bf16* __restrict__ w2b,
                  const bf16* __restrict__ w3b,
                  const int* __restrict__ cnt, const int* __restrict__ list,
                  const float* __restrict__ wlist,
                  float* __restrict__ out) {
  const int e = blockIdx.x;
  const int tilebase = blockIdx.y * ROWS;
  const int ce = cnt[e];
  if (tilebase >= ce) return;                               // uniform per block

  extern __shared__ char smem[];
  bf16*  xs   = (bf16*)smem;                                // ROWS x LSTR @ LDS offset 0
  bf16*  hs   = (bf16*)(smem + (size_t)ROWS * LSTR * sizeof(bf16));
  int*   stok = (int*)(smem + (size_t)2 * ROWS * LSTR * sizeof(bf16));
  float* swt  = (float*)(stok + ROWS);

  const int tid = threadIdx.x, lane = tid & 31, wid = tid >> 5;
  const int rb = wid & 1, cg = wid >> 1;
  const int rowbase = rb * 16;

  if (tid < ROWS) {
    int g = tilebase + tid;
    int t = 0; float w = 0.f;
    if (g < ce) { t = list[(size_t)e * T_TOK + g]; w = wlist[(size_t)e * T_TOK + g]; }
    stok[tid] = t; swt[tid] = w;
  }
  __syncthreads();

  // stage 0: async-gather bf16 token rows -> LDS
  for (int idx = tid; idx < ROWS * (DIM / 8); idx += 256) {
    int m  = idx >> 7;
    int d8 = (idx & 127) << 3;
    async_b128_to_lds((unsigned)(m * LSTR + d8) * 2u,
                      xb + (size_t)stok[m] * DIM + d8);
  }
  wait_async_all();
  __syncthreads();

  const bf16* w1e = w1b + (size_t)e * IDIM * DIM;
  const bf16* w3e = w3b + (size_t)e * IDIM * DIM;
  const bf16* w2e = w2b + (size_t)e * DIM * IDIM;
  const v8f vz = {0.f,0.f,0.f,0.f,0.f,0.f,0.f,0.f};

  // stage 1: h = silu(x w1^T) * (x w3^T) -> LDS
#pragma unroll 1
  for (int nt = 0; nt < 16; ++nt) {
    const int ntile = cg * 16 + nt;                         // inter ntile
    v8f c1 = vz, c3 = vz;
#pragma unroll 2
    for (int kc = 0; kc < DIM / 32; ++kc) {
      v16bf a  = load_a_frag(xs, LSTR, rowbase, kc << 5, lane);
      v16bf b1 = load_b_tiled(w1e, DIM / 32, ntile, kc, lane);
      v16bf b3 = load_b_tiled(w3e, DIM / 32, ntile, kc, lane);
      c1 = WMMA_BF16(a, b1, c1);
      c3 = WMMA_BF16(a, b3, c3);
    }
    const int mbase = rowbase + ((lane >> 4) << 3);
    const int nloc  = ntile * 16 + (lane & 15);
#pragma unroll
    for (int r = 0; r < 8; ++r) {
      float s = c1[r];
      float h = (s / (1.f + __expf(-s))) * c3[r];
      hs[(mbase + r) * LSTR + nloc] = f2bf(h);
    }
  }
  __syncthreads();

  // stage 2: y_contrib = weight * (h w2^T), atomic combine into out
#pragma unroll 1
  for (int nt = 0; nt < 16; ++nt) {
    const int ntile = cg * 16 + nt;                         // output ntile in D
    v8f c = vz;
#pragma unroll 2
    for (int kc = 0; kc < IDIM / 32; ++kc) {
      v16bf a = load_a_frag(hs, LSTR, rowbase, kc << 5, lane);
      v16bf b = load_b_tiled(w2e, IDIM / 32, ntile, kc, lane);
      c = WMMA_BF16(a, b, c);
    }
    const int mbase = rowbase + ((lane >> 4) << 3);
    const int n = ntile * 16 + (lane & 15);
#pragma unroll
    for (int r = 0; r < 8; ++r) {
      const int m = mbase + r;
      if (tilebase + m < ce)
        atomicAdd(out + (size_t)stok[m] * DIM + n, swt[m] * c[r]);
    }
  }
}

// ---------------- launcher ----------------
extern "C" void kernel_launch(void* const* d_in, const int* in_sizes, int n_in,
                              void* d_out, int out_size, void* d_ws, size_t ws_size,
                              hipStream_t stream) {
  const float* x   = (const float*)d_in[0];
  const float* gw  = (const float*)d_in[1];
  const float* w1  = (const float*)d_in[2];
  const float* w2  = (const float*)d_in[3];
  const float* w3  = (const float*)d_in[4];
  const float* sw1 = (const float*)d_in[5];
  const float* sb1 = (const float*)d_in[6];
  const float* sw2 = (const float*)d_in[7];
  const float* sb2 = (const float*)d_in[8];
  const float* sw3 = (const float*)d_in[9];
  const float* sb3 = (const float*)d_in[10];
  float* out = (float*)d_out;

  char* ws = (char*)d_ws;
  // control block
  int*   cnt  = (int*)ws;                                       // E ints
  float* psum = (float*)(ws + 256);                             // E floats
  int*   list = (int*)(ws + 512);                               // E*T ints   (512KB)
  float* wl   = (float*)(ws + 512 + (size_t)NEXP * T_TOK * 4);  // E*T floats (512KB)
  // bf16 staging area (fragment-tiled weights + row-major x)
  size_t off = 1052672;                                         // 1MB+4KB, 4KB aligned
  bf16* xbuf  = (bf16*)(ws + off); off += (size_t)T_TOK * DIM * 2;        // 16MB
  bf16* w1b   = (bf16*)(ws + off); off += (size_t)NEXP * IDIM * DIM * 2;  // 32MB
  bf16* w3b   = (bf16*)(ws + off); off += (size_t)NEXP * IDIM * DIM * 2;  // 32MB
  bf16* w2b   = (bf16*)(ws + off); off += (size_t)NEXP * DIM * IDIM * 2;  // 32MB
  bf16* sw1b  = (bf16*)(ws + off); off += (size_t)SHI * DIM * 2;          // 4MB
  bf16* sw3b  = (bf16*)(ws + off); off += (size_t)SHI * DIM * 2;          // 4MB
  bf16* sw2b  = (bf16*)(ws + off); off += (size_t)DIM * SHI * 2;          // 4MB

  moe_init_kernel<<<dim3(1), dim3(64), 0, stream>>>(cnt, psum);
  moe_gate_kernel<<<dim3(T_TOK / 256), dim3(256), 0, stream>>>(x, gw, cnt, psum, list, wl);
  moe_aux_kernel<<<dim3(1), dim3(32), 0, stream>>>(cnt, psum, out + (size_t)T_TOK * DIM);

  // bf16 staging (deterministic, runs every call)
  {
    long t8;
    t8 = (long)T_TOK * DIM / 8;
    convert_x_kernel<<<dim3((unsigned)(t8 / 256)), dim3(256), 0, stream>>>(x, xbuf);
    t8 = (long)NEXP * IDIM * DIM / 8;   // K=1024 -> ksh=5
    convert_w_kernel<<<dim3((unsigned)((t8 + 255) / 256)), dim3(256), 0, stream>>>(w1, w1b, DIM, 5, t8);
    convert_w_kernel<<<dim3((unsigned)((t8 + 255) / 256)), dim3(256), 0, stream>>>(w3, w3b, DIM, 5, t8);
    convert_w_kernel<<<dim3((unsigned)((t8 + 255) / 256)), dim3(256), 0, stream>>>(w2, w2b, IDIM, 5, t8);
    t8 = (long)SHI * DIM / 8;
    convert_w_kernel<<<dim3((unsigned)((t8 + 255) / 256)), dim3(256), 0, stream>>>(sw1, sw1b, DIM, 5, t8);
    convert_w_kernel<<<dim3((unsigned)((t8 + 255) / 256)), dim3(256), 0, stream>>>(sw3, sw3b, DIM, 5, t8);
    t8 = (long)DIM * SHI / 8;           // K=2048 -> ksh=6
    convert_w_kernel<<<dim3((unsigned)((t8 + 255) / 256)), dim3(256), 0, stream>>>(sw2, sw2b, SHI, 6, t8);
  }

  const size_t sh_smem = (size_t)2 * ROWS * LSTR * sizeof(bf16);          // ~129 KB
  moe_shared_kernel<<<dim3(T_TOK / ROWS), dim3(256), sh_smem, stream>>>(
      xbuf, sw1b, sb1, sw2b, sb2, sw3b, sb3, out);

  const size_t rt_smem = sh_smem + (size_t)ROWS * 8;                      // + tok/wt
  moe_routed_kernel<<<dim3(NEXP, T_TOK / ROWS), dim3(256), rt_smem, stream>>>(
      xbuf, w1b, w2b, w3b, cnt, list, wl, out);
}
